// MultiHeadedAttention_81003083203502
// MI455X (gfx1250) — matrix operand order repaired and checked by reading, not compiled, not run on hardware
//
#include <hip/hip_runtime.h>
#include <math.h>

// Problem constants (from reference)
constexpr int EMB   = 1024;
constexpr int HEADS = 16;
constexpr int HEADD = 64;   // EMB / HEADS
constexpr int BB    = 4;
constexpr int SS    = 1024;

typedef __bf16 bf16_t;
typedef __bf16 v16bf __attribute__((ext_vector_type(16)));
typedef float  v8f   __attribute__((ext_vector_type(8)));

__device__ __forceinline__ v8f wmma_bf16(v16bf a, v16bf b, v8f c) {
  // D = A(16x32 bf16) x B(32x16 bf16) + C(16x16 f32)
  return __builtin_amdgcn_wmma_f32_16x16x32_bf16(
      /*neg_a=*/false, a, /*neg_b=*/false, b,
      /*c_mod=*/(short)0, c, /*reuse_a=*/false, /*reuse_b=*/false);
}

// ---------------- fragment loaders (wave32 layouts per CDNA5 ISA 7.12.2) ----
// A 16x32 bf16: lane L -> row m=L%16; half h=L/16; VGPR v<4 : K=2v+8h (pair),
//               VGPR v>=4: K=16+2(v-4)+8h (pair). Pairs contiguous along K.
__device__ __forceinline__ v16bf load_a32_f32(const float* base, int lda, int lane) {
  const int m = lane & 15, h = lane >> 4;
  const float* row = base + (size_t)m * lda + 8 * h;
  v16bf a;
#pragma unroll
  for (int v = 0; v < 8; ++v) {
    const int k = (v < 4) ? (2 * v) : (8 + 2 * v);  // (+8h folded into row ptr)
    a[2 * v]     = (bf16_t)row[k];
    a[2 * v + 1] = (bf16_t)row[k + 1];
  }
  return a;
}

__device__ __forceinline__ v16bf load_a32_bf16(const bf16_t* base, int lda, int lane) {
  const int m = lane & 15, h = lane >> 4;
  const bf16_t* row = base + (size_t)m * lda + 8 * h;
  v16bf a;
#pragma unroll
  for (int v = 0; v < 8; ++v) {
    const int k = (v < 4) ? (2 * v) : (8 + 2 * v);
    a[2 * v]     = row[k];
    a[2 * v + 1] = row[k + 1];
  }
  return a;
}

// B 32x16 bf16 from a transposed (NxK row-major) bf16 matrix: reduction contiguous.
// Lane L -> col n=L%16; half h; VGPR v<4: K=2v+8h pair, v>=4: K=16+2(v-4)+8h pair.
__device__ __forceinline__ v16bf load_bT_bf16(const bf16_t* baseT, int ldt, int lane) {
  const int n = lane & 15, h = lane >> 4;
  const bf16_t* row = baseT + (size_t)n * ldt + 8 * h;
  v16bf b;
#pragma unroll
  for (int v = 0; v < 8; ++v) {
    const int k = (v < 4) ? (2 * v) : (8 + 2 * v);
    b[2 * v]     = row[k];
    b[2 * v + 1] = row[k + 1];
  }
  return b;
}

// ---------------- prep: weight transpose + f32->bf16 ----------------------
// Wt[which][h][n][e] = W_which[h][e][n]   (reduction dim e contiguous)
__global__ __launch_bounds__(256) void pack_qkv_w_kernel(
    const float* __restrict__ Wk, const float* __restrict__ Wv,
    const float* __restrict__ Wq, bf16_t* __restrict__ Wt) {
  const size_t idx = (size_t)blockIdx.x * blockDim.x + threadIdx.x;
  const int e = (int)(idx % EMB);
  size_t t = idx / EMB;
  const int n = (int)(t % HEADD);
  t /= HEADD;
  const int h = (int)(t % HEADS);
  const int which = (int)(t / HEADS);
  const float* W = (which == 0) ? Wk : (which == 1) ? Wv : Wq;
  Wt[idx] = (bf16_t)W[((size_t)h * EMB + e) * HEADD + n];
}

// Wo_t[n][e] = Wo[e][n]
__global__ __launch_bounds__(256) void pack_wo_kernel(
    const float* __restrict__ Wo, bf16_t* __restrict__ Wo_t) {
  const size_t idx = (size_t)blockIdx.x * blockDim.x + threadIdx.x;
  const int e = (int)(idx % EMB);
  const int n = (int)(idx / EMB);
  Wo_t[idx] = (bf16_t)Wo[(size_t)e * EMB + n];
}

// ---------------- kernel 1: QKV projections -------------------------------
// grid = (S/32, HEADS, B*3), block = 32 (one wave). Each wave: 32x64 tile,
// so every B fragment feeds two WMMAs.
__global__ __launch_bounds__(32) void qkv_proj_kernel(
    const float* __restrict__ Xk, const float* __restrict__ Xv, const float* __restrict__ Xq,
    const bf16_t* __restrict__ Wt,
    bf16_t* __restrict__ Kb, bf16_t* __restrict__ Vt, bf16_t* __restrict__ Qb) {
  const int st = blockIdx.x;          // 32-row tile index
  const int h  = blockIdx.y;
  const int b  = blockIdx.z / 3;
  const int which = blockIdx.z % 3;
  const int lane = threadIdx.x & 31;

  const float* X = (which == 0) ? Xk : (which == 1) ? Xv : Xq;
  const float* Xt0 = X + ((size_t)b * SS + (size_t)st * 32) * EMB;
  const float* Xt1 = Xt0 + (size_t)16 * EMB;
  const bf16_t* Wh = Wt + (size_t)(which * HEADS + h) * HEADD * EMB;

  const v8f vzero = {0.f, 0.f, 0.f, 0.f, 0.f, 0.f, 0.f, 0.f};
  v8f acc[2][4];
#pragma unroll
  for (int i = 0; i < 2; ++i)
#pragma unroll
    for (int j = 0; j < 4; ++j) acc[i][j] = vzero;

  for (int e = 0; e < EMB; e += 32) {
    v16bf a0 = load_a32_f32(Xt0 + e, EMB, lane);
    v16bf a1 = load_a32_f32(Xt1 + e, EMB, lane);
#pragma unroll
    for (int j = 0; j < 4; ++j) {
      v16bf bm = load_bT_bf16(Wh + (size_t)(16 * j) * EMB + e, EMB, lane);
      acc[0][j] = wmma_bf16(a0, bm, acc[0][j]);
      acc[1][j] = wmma_bf16(a1, bm, acc[1][j]);
    }
  }

  const int half = lane >> 4, n = lane & 15;
  if (which == 1) {
    // V stored transposed: [b,h][d][s]  (reduction dim s contiguous for PV)
    bf16_t* dst = Vt + (size_t)(b * HEADS + h) * HEADD * SS;
#pragma unroll
    for (int i = 0; i < 2; ++i)
#pragma unroll
      for (int j = 0; j < 4; ++j)
#pragma unroll
        for (int r = 0; r < 8; ++r) {
          const int s = st * 32 + 16 * i + r + 8 * half;
          const int d = 16 * j + n;
          dst[(size_t)d * SS + s] = (bf16_t)acc[i][j][r];
        }
  } else {
    // Q, K stored row-major: [b,h][s][d]  (d contiguous for QK^T)
    bf16_t* dst = ((which == 0) ? Kb : Qb) +
                  ((size_t)(b * HEADS + h) * SS + (size_t)st * 32) * HEADD;
#pragma unroll
    for (int i = 0; i < 2; ++i)
#pragma unroll
      for (int j = 0; j < 4; ++j)
#pragma unroll
        for (int r = 0; r < 8; ++r)
          dst[(size_t)(16 * i + r + 8 * half) * HEADD + 16 * j + n] =
              (bf16_t)acc[i][j][r];
  }
}

// ---------------- kernel 2: flash-style causal attention ------------------
// grid = (S/16, HEADS, B), block = 32. Each wave: one 16-query tile, online softmax.
__global__ __launch_bounds__(32) void attn_kernel(
    const bf16_t* __restrict__ Qb, const bf16_t* __restrict__ Kb,
    const bf16_t* __restrict__ Vt, bf16_t* __restrict__ Ctx) {
  __shared__ bf16_t Pst[16 * 32];  // staged P tile (C-layout -> A-layout reshape)

  const int qt = blockIdx.x, h = blockIdx.y, b = blockIdx.z;
  const int lane = threadIdx.x & 31;
  const int half = lane >> 4, n = lane & 15;
  const float NEG_INF = -__builtin_inff();
  const float scale = 0.03125f;  // 1/sqrt(S=1024); ref scales (logits+mask)

  const bf16_t* Qh = Qb + ((size_t)(b * HEADS + h) * SS + (size_t)qt * 16) * HEADD;
  const bf16_t* Kh = Kb + (size_t)(b * HEADS + h) * SS * HEADD;
  const bf16_t* Vh = Vt + (size_t)(b * HEADS + h) * HEADD * SS;

  const v16bf qa0 = load_a32_bf16(Qh, HEADD, lane);       // d = 0..31
  const v16bf qa1 = load_a32_bf16(Qh + 32, HEADD, lane);  // d = 32..63

  const v8f vzero = {0.f, 0.f, 0.f, 0.f, 0.f, 0.f, 0.f, 0.f};
  v8f acc[4];
#pragma unroll
  for (int j = 0; j < 4; ++j) acc[j] = vzero;
  float mrun[8], lrun[8];
#pragma unroll
  for (int r = 0; r < 8; ++r) { mrun[r] = NEG_INF; lrun[r] = 0.f; }

  for (int k0 = 0; k0 <= qt * 16; k0 += 32) {
    // logits for 32 keys = two 16-wide C tiles
    v8f lg0 = vzero, lg1 = vzero;
    {
      v16bf b0 = load_bT_bf16(Kh + (size_t)k0 * HEADD, HEADD, lane);
      v16bf b1 = load_bT_bf16(Kh + (size_t)k0 * HEADD + 32, HEADD, lane);
      lg0 = wmma_bf16(qa0, b0, lg0);
      lg0 = wmma_bf16(qa1, b1, lg0);
    }
    if (k0 + 16 <= qt * 16 + 15) {  // second tile has any unmasked key
      v16bf b0 = load_bT_bf16(Kh + (size_t)(k0 + 16) * HEADD, HEADD, lane);
      v16bf b1 = load_bT_bf16(Kh + (size_t)(k0 + 16) * HEADD + 32, HEADD, lane);
      lg1 = wmma_bf16(qa0, b0, lg1);
      lg1 = wmma_bf16(qa1, b1, lg1);
    }

    // online softmax, one row per (VGPR, half): m = r + 8*half
#pragma unroll
    for (int r = 0; r < 8; ++r) {
      const int m = r + 8 * half;
      const int q = qt * 16 + m;
      float x0 = (k0 + n <= q)      ? lg0[r] * scale : NEG_INF;
      float x1 = (k0 + 16 + n <= q) ? lg1[r] * scale : NEG_INF;
      float rmax = fmaxf(x0, x1);
#pragma unroll
      for (int off = 8; off >= 1; off >>= 1)
        rmax = fmaxf(rmax, __shfl_xor(rmax, off, 32));
      const float mn = fmaxf(mrun[r], rmax);        // finite: key k0 always valid
      const float corr = __expf(mrun[r] - mn);      // 0 on first iteration
      const float p0 = __expf(x0 - mn);
      const float p1 = __expf(x1 - mn);
      float rsum = p0 + p1;
#pragma unroll
      for (int off = 8; off >= 1; off >>= 1)
        rsum += __shfl_xor(rsum, off, 32);
      lrun[r] = lrun[r] * corr + rsum;
      mrun[r] = mn;
#pragma unroll
      for (int j = 0; j < 4; ++j) acc[j][r] = acc[j][r] * corr;
      Pst[m * 32 + n]      = (bf16_t)p0;
      Pst[m * 32 + 16 + n] = (bf16_t)p1;
    }
    __syncthreads();  // single-wave WG: DS in-order anyway; keeps IR honest

    // acc += P(16x32) x V(32x64)
    const v16bf pa = load_a32_bf16(Pst, 32, lane);
#pragma unroll
    for (int j = 0; j < 4; ++j) {
      v16bf vb = load_bT_bf16(Vh + (size_t)(16 * j) * SS + k0, SS, lane);
      acc[j] = wmma_bf16(pa, vb, acc[j]);
    }
    __syncthreads();
  }

  // normalize and write concat buffer [b][s][h*64+d] (A-ready for out proj)
  float inv[8];
#pragma unroll
  for (int r = 0; r < 8; ++r) inv[r] = 1.0f / lrun[r];
  bf16_t* dst = Ctx + ((size_t)b * SS + (size_t)qt * 16) * EMB + h * HEADD;
#pragma unroll
  for (int j = 0; j < 4; ++j)
#pragma unroll
    for (int r = 0; r < 8; ++r)
      dst[(size_t)(r + 8 * half) * EMB + 16 * j + n] = (bf16_t)(acc[j][r] * inv[r]);
}

// ---------------- kernel 3: output projection + bias + ReLU ---------------
// grid = ((B*S)/32, EMB/64), block = 32. Each wave: 32x64 tile.
__global__ __launch_bounds__(32) void outproj_kernel(
    const bf16_t* __restrict__ Ctx, const bf16_t* __restrict__ Wo_t,
    const float* __restrict__ bo, float* __restrict__ Out) {
  const int rt = blockIdx.x;
  const int nt = blockIdx.y;
  const int lane = threadIdx.x & 31;
  const bf16_t* A0 = Ctx + (size_t)rt * 32 * EMB;
  const bf16_t* A1 = A0 + (size_t)16 * EMB;

  const v8f vzero = {0.f, 0.f, 0.f, 0.f, 0.f, 0.f, 0.f, 0.f};
  v8f acc[2][4];
#pragma unroll
  for (int i = 0; i < 2; ++i)
#pragma unroll
    for (int j = 0; j < 4; ++j) acc[i][j] = vzero;

  for (int e = 0; e < EMB; e += 32) {
    v16bf a0 = load_a32_bf16(A0 + e, EMB, lane);
    v16bf a1 = load_a32_bf16(A1 + e, EMB, lane);
#pragma unroll
    for (int j = 0; j < 4; ++j) {
      v16bf bm = load_bT_bf16(Wo_t + (size_t)(nt * 64 + 16 * j) * EMB + e, EMB, lane);
      acc[0][j] = wmma_bf16(a0, bm, acc[0][j]);
      acc[1][j] = wmma_bf16(a1, bm, acc[1][j]);
    }
  }

  const int half = lane >> 4, n = lane & 15;
#pragma unroll
  for (int i = 0; i < 2; ++i)
#pragma unroll
    for (int j = 0; j < 4; ++j)
#pragma unroll
      for (int r = 0; r < 8; ++r) {
        const int row = rt * 32 + 16 * i + r + 8 * half;
        const int col = nt * 64 + 16 * j + n;
        const float v = acc[i][j][r] + bo[col];
        Out[(size_t)row * EMB + col] = fmaxf(v, 0.0f);
      }
}

// ---------------- launch ---------------------------------------------------
extern "C" void kernel_launch(void* const* d_in, const int* in_sizes, int n_in,
                              void* d_out, int out_size, void* d_ws, size_t ws_size,
                              hipStream_t stream) {
  const float* Xk = (const float*)d_in[0];
  const float* Xv = (const float*)d_in[1];
  const float* Xq = (const float*)d_in[2];
  const float* Wk = (const float*)d_in[3];
  const float* Wv = (const float*)d_in[4];
  const float* Wq = (const float*)d_in[5];
  const float* Wo = (const float*)d_in[6];
  const float* bo = (const float*)d_in[7];
  float* Out = (float*)d_out;

  // workspace: Q, K, Vt, Ctx (4 x 8 MB) + Wt (6 MB) + Wo_t (2 MB), all bf16
  const size_t nTok = (size_t)BB * HEADS * SS * HEADD;   // == B*S*EMB
  const size_t nWt  = (size_t)3 * HEADS * HEADD * EMB;
  const size_t nWo  = (size_t)EMB * EMB;
  bf16_t* Qb   = (bf16_t*)d_ws;
  bf16_t* Kb   = Qb + nTok;
  bf16_t* Vt   = Kb + nTok;
  bf16_t* Ctx  = Vt + nTok;
  bf16_t* Wt   = Ctx + nTok;
  bf16_t* Wo_t = Wt + nWt;

  const dim3 blk(32, 1, 1);
  pack_qkv_w_kernel<<<dim3((unsigned)(nWt / 256)), dim3(256), 0, stream>>>(Wk, Wv, Wq, Wt);
  pack_wo_kernel<<<dim3((unsigned)(nWo / 256)), dim3(256), 0, stream>>>(Wo, Wo_t);
  qkv_proj_kernel<<<dim3(SS / 32, HEADS, BB * 3), blk, 0, stream>>>(
      Xk, Xv, Xq, Wt, Kb, Vt, Qb);
  attn_kernel<<<dim3(SS / 16, HEADS, BB), blk, 0, stream>>>(Qb, Kb, Vt, Ctx);
  outproj_kernel<<<dim3((BB * SS) / 32, EMB / 64, 1), blk, 0, stream>>>(
      Ctx, Wo_t, bo, Out);
}
